// ContextPythiaModel_38878043963860
// MI455X (gfx1250) — compile-verified, matrix operand order checked by reading, not run on hardware
//
#include <hip/hip_runtime.h>
#include <hip/hip_bf16.h>

// ---------------------------------------------------------------------------
// ContextPythia forward for MI455X (gfx1250, wave32):
//   - bf16 WMMA 16x16x32 for all GEMMs / attention
//   - B panels staged via GLOBAL_LOAD_ASYNC_TO_LDS_B128 (ASYNCcnt)
//   - A panels staged via Tensor Data Mover (TENSOR_LOAD_TO_LDS, TENSORcnt)
// ---------------------------------------------------------------------------
typedef __attribute__((ext_vector_type(16))) __bf16 v16bf;
typedef __attribute__((ext_vector_type(8)))  __bf16 v8bf;
typedef __attribute__((ext_vector_type(8)))  float  v8f;
typedef __attribute__((ext_vector_type(4)))  unsigned int u32x4;
typedef __attribute__((ext_vector_type(8)))  int i32x8;
typedef __attribute__((ext_vector_type(4)))  int i32x4;

#define LN_EPS 1e-5f

#if defined(__has_builtin)
#  if __has_builtin(__builtin_amdgcn_tensor_load_to_lds) && \
      __has_builtin(__builtin_amdgcn_s_wait_tensorcnt)
#    define USE_TDM 1
#  endif
#endif
#ifndef USE_TDM
#  define USE_TDM 0
#endif

// ---- CDNA5 async global->LDS (ISA 15.18.3 op 98) ---------------------------
__device__ __forceinline__ void async_copy_b128(const void* gsrc, void* ldst) {
  const uint32_t l = (uint32_t)(uintptr_t)ldst;      // low 32 bits == LDS offset
  const uint64_t g = (uint64_t)(uintptr_t)gsrc;
  asm volatile("global_load_async_to_lds_b128 %0, %1, off"
               :: "v"(l), "v"(g) : "memory");
}
__device__ __forceinline__ void wait_async0() {
  asm volatile("s_wait_asynccnt 0" ::: "memory");
}

#if USE_TDM
// ---- Tensor Data Mover: 2D tile (rows x 32 bf16 elems), row stride K -------
// D# per ISA ch.8: group0 {count=1, lds_addr, global_addr, type=2},
// group1 {data_size=1(2B), tensor_dim0=K, tensor_dim1=rows, tile_dim0=32,
//         tile_dim1=rows, tensor_dim0_stride=K}, groups 2/3 zero (2D tensor).
// This toolchain exposes the 6-arg builtin (extra zero-filled i32x8 group).
__device__ __forceinline__ void tdm_load_tile(uint32_t lds_addr, const void* g,
                                              uint32_t K_elems, uint32_t rows) {
  const uint64_t ga = (uint64_t)(uintptr_t)g;
  u32x4 g0;
  g0[0] = 1u;                                               // count=1 (valid)
  g0[1] = lds_addr;                                         // LDS byte address
  g0[2] = (uint32_t)ga;                                     // global_addr lo
  g0[3] = (uint32_t)((ga >> 32) & 0x01FFFFFFu) | (2u << 30);// addr hi | type=2
  i32x8 g1;
  g1[0] = (int)(1u << 16);                                  // data_size=2B
  g1[1] = (int)((K_elems & 0xFFFFu) << 16);                 // tensor_dim0 lo16
  g1[2] = (int)(((K_elems >> 16) & 0xFFFFu) |
                ((rows & 0xFFFFu) << 16));                  // dim0 hi | dim1 lo
  g1[3] = (int)(((rows >> 16) & 0xFFFFu) | (32u << 16));    // dim1 hi | tile_dim0=32
  g1[4] = (int)(rows & 0xFFFFu);                            // tile_dim1=rows
  g1[5] = (int)K_elems;                                     // dim0_stride lo32
  g1[6] = 0;                                                // stride hi | dim1_stride
  g1[7] = 0;
  const i32x4 z4 = {0, 0, 0, 0};
  const i32x8 z8 = {0, 0, 0, 0, 0, 0, 0, 0};
  __builtin_amdgcn_tensor_load_to_lds(g0, g1, z4, z4, z8, 0);
}
#endif

// --------------------------- block-wide sum (256 thr) ----------------------
__device__ __forceinline__ float block_sum_256(float v, float* red) {
#pragma unroll
  for (int off = 16; off > 0; off >>= 1) v += __shfl_xor(v, off, 32);
  const int wave = threadIdx.x >> 5;
  const int lane = threadIdx.x & 31;
  if (lane == 0) red[wave] = v;
  __syncthreads();
  float s = (wave == 0 && lane < 8) ? red[lane] : 0.0f;
  if (wave == 0) {
#pragma unroll
    for (int off = 4; off > 0; off >>= 1) s += __shfl_xor(s, off, 32);
    if (lane == 0) red[0] = s;
  }
  __syncthreads();
  const float r = red[0];
  __syncthreads();
  return r;
}

// --------------------------- embedding gather + LN -> bf16 ------------------
__global__ void __launch_bounds__(256) embed_ln_kernel(
    const int* __restrict__ ids, const float* __restrict__ W,
    const float* __restrict__ g, const float* __restrict__ beta,
    __bf16* __restrict__ out) {
  __shared__ float red[32];
  const int row = blockIdx.x;            // 0..2047 (= b*512+s)
  const int t   = threadIdx.x;
  const float* src = W + (size_t)ids[row] * 512;
  const float v0 = src[t];
  const float v1 = src[t + 256];
  const float mean = block_sum_256(v0 + v1, red) * (1.0f / 512.0f);
  const float d0 = v0 - mean, d1 = v1 - mean;
  const float var = block_sum_256(d0 * d0 + d1 * d1, red) * (1.0f / 512.0f);
  const float inv = rsqrtf(var + LN_EPS);
  out[(size_t)row * 512 + t]       = (__bf16)(d0 * inv * g[t] + beta[t]);
  out[(size_t)row * 512 + t + 256] = (__bf16)(d1 * inv * g[t + 256] + beta[t + 256]);
}

// --------------------------- LN over H=512 -> bf16 --------------------------
__global__ void __launch_bounds__(256) ln512_kernel(
    const float* __restrict__ x, const float* __restrict__ g,
    const float* __restrict__ beta, __bf16* __restrict__ out) {
  __shared__ float red[32];
  const int row = blockIdx.x;
  const int t   = threadIdx.x;
  const float* xp = x + (size_t)row * 512;
  const float v0 = xp[t];
  const float v1 = xp[t + 256];
  const float mean = block_sum_256(v0 + v1, red) * (1.0f / 512.0f);
  const float d0 = v0 - mean, d1 = v1 - mean;
  const float var = block_sum_256(d0 * d0 + d1 * d1, red) * (1.0f / 512.0f);
  const float inv = rsqrtf(var + LN_EPS);
  out[(size_t)row * 512 + t]       = (__bf16)(d0 * inv * g[t] + beta[t]);
  out[(size_t)row * 512 + t + 256] = (__bf16)(d1 * inv * g[t + 256] + beta[t + 256]);
}

// --------------------------- weight pack f32 -> bf16 WMMA-B layout ----------
__global__ void pack_b_kernel(const float* __restrict__ W, __bf16* __restrict__ Bp,
                              int K, int N) {
  const int Kt = K >> 5;
  const size_t total = (size_t)K * N;
  for (size_t idx = (size_t)blockIdx.x * blockDim.x + threadIdx.x; idx < total;
       idx += (size_t)gridDim.x * blockDim.x) {
    const size_t tile = idx >> 9;
    const int e    = (int)(idx & 511);
    const int lane = e >> 4;
    const int i    = e & 15;
    const int nl   = lane & 15;
    const int half = lane >> 4;
    const int kt   = (int)(tile % Kt);
    const int nt   = (int)(tile / Kt);
    const int kl   = (i < 8) ? (half * 8 + i) : (16 + half * 8 + (i - 8));
    const int k    = kt * 32 + kl;
    const int n    = nt * 16 + nl;
    Bp[idx] = (__bf16)W[(size_t)k * N + n];
  }
}

// --------------------------- bf16 WMMA GEMM ---------------------------------
// C = A(MxK)*B(KxN) + bias.  Block tile 128x64, 8 waves, wave w owns rows
// [16w,16w+16) and all 64 block columns (4 accumulators share one A fragment).
// Per K-step: A panel (128x32 bf16, 8KB) via TDM; B panel (4KB, pre-packed
// lane layout) via per-thread async b128.  Both double buffered.
template <int ACT, int OUT_BF16>
__global__ void __launch_bounds__(256) gemm_bf16_wmma(
    const __bf16* __restrict__ A, const __bf16* __restrict__ Bp,
    const float* __restrict__ bias, void* __restrict__ Cout,
    int M, int N, int K) {
  (void)M;
  __shared__ __bf16 Bs[2][4 * 512];  // two 4KB B panels
#if USE_TDM
  __shared__ __bf16 As[2][128 * 32]; // two 8KB A panels
#endif
  const int tid  = threadIdx.x;
  const int lane = tid & 31;
  const int wave = tid >> 5;
  const int n16  = lane & 15;
  const int half = lane >> 4;
  const int tileM = blockIdx.y * 128 + wave * 16;
  const int tileN = blockIdx.x * 64;
  const int Kt = K >> 5;
  const int nt0 = tileN >> 4;

  const int cp_ntl   = tid >> 6;
  const int cp_chunk = tid & 63;
  const __bf16* bsrc0 = Bp + ((size_t)(nt0 + cp_ntl) * Kt) * 512 + cp_chunk * 8;
  __bf16* bdst[2] = {&Bs[0][cp_ntl * 512 + cp_chunk * 8],
                     &Bs[1][cp_ntl * 512 + cp_chunk * 8]};

#if USE_TDM
  const __bf16* apanel = A + (size_t)(blockIdx.y * 128) * K;  // block's A rows
#else
  const __bf16* ap = A + (size_t)(tileM + n16) * K + half * 8;
#endif

  v8f acc[4];
  const v8f zero = {0.f, 0.f, 0.f, 0.f, 0.f, 0.f, 0.f, 0.f};
#pragma unroll
  for (int t = 0; t < 4; ++t) acc[t] = zero;

#if USE_TDM
  if (wave == 0)
    tdm_load_tile((uint32_t)(uintptr_t)&As[0][0], apanel, (uint32_t)K, 128u);
#endif
  async_copy_b128(bsrc0, bdst[0]);

  for (int kt = 0; kt < Kt; ++kt) {
#if USE_TDM
    if (wave == 0) __builtin_amdgcn_s_wait_tensorcnt(0);
#endif
    wait_async0();
    __syncthreads();                 // panels for kt visible to all waves
    if (kt + 1 < Kt) {               // overlap next panels with compute
#if USE_TDM
      if (wave == 0)
        tdm_load_tile((uint32_t)(uintptr_t)&As[(kt + 1) & 1][0],
                      apanel + (size_t)(kt + 1) * 32, (uint32_t)K, 128u);
#endif
      async_copy_b128(bsrc0 + (size_t)(kt + 1) * 512, bdst[(kt + 1) & 1]);
    }

#if USE_TDM
    const __bf16* as = &As[kt & 1][(wave * 16 + n16) * 32];
    const v8bf alo = *reinterpret_cast<const v8bf*>(as + half * 8);
    const v8bf ahi = *reinterpret_cast<const v8bf*>(as + 16 + half * 8);
#else
    const v8bf alo = *reinterpret_cast<const v8bf*>(ap);
    const v8bf ahi = *reinterpret_cast<const v8bf*>(ap + 16);
    ap += 32;
#endif
    const v16bf a = __builtin_shufflevector(alo, ahi, 0, 1, 2, 3, 4, 5, 6, 7,
                                            8, 9, 10, 11, 12, 13, 14, 15);
    const __bf16* bs = &Bs[kt & 1][lane * 16];
#pragma unroll
    for (int t = 0; t < 4; ++t) {
      const v8bf blo = *reinterpret_cast<const v8bf*>(bs + t * 512);
      const v8bf bhi = *reinterpret_cast<const v8bf*>(bs + t * 512 + 8);
      const v16bf b = __builtin_shufflevector(blo, bhi, 0, 1, 2, 3, 4, 5, 6, 7,
                                              8, 9, 10, 11, 12, 13, 14, 15);
      acc[t] = __builtin_amdgcn_wmma_f32_16x16x32_bf16(false, a, false, b,
                                                       (short)0, acc[t],
                                                       false, false);
    }
  }

#pragma unroll
  for (int t = 0; t < 4; ++t) {
    const int col = tileN + t * 16 + n16;
    const float bv = bias ? bias[col] : 0.0f;
#pragma unroll
    for (int r = 0; r < 8; ++r) {
      const int row = tileM + half * 8 + r;
      float v = acc[t][r] + bv;
      if (ACT == 1) v = 0.5f * v * (1.0f + erff(v * 0.70710678118f));  // GELU
      if (OUT_BF16)
        reinterpret_cast<__bf16*>(Cout)[(size_t)row * N + col] = (__bf16)v;
      else
        reinterpret_cast<float*>(Cout)[(size_t)row * N + col] = v;
    }
  }
}

// --------------------------- serial context scan ----------------------------
__global__ void __launch_bounds__(256) scan_kernel(
    const float* __restrict__ cbW, const float* __restrict__ embproj,
    const float* __restrict__ prev, const float* __restrict__ g,
    const float* __restrict__ beta, __bf16* __restrict__ ctxout) {
  __shared__ float ctx[256];
  __shared__ float red[32];
  const int b = blockIdx.x;
  const int j = threadIdx.x;
  float w[256];
#pragma unroll
  for (int k = 0; k < 256; ++k) w[k] = cbW[k * 256 + j];  // W_c = cb_W[0:256,:]
  ctx[j] = prev[b * 256 + j];
  const float gj = g[j];
  const float bj = beta[j];
  __syncthreads();
  for (int t = 0; t < 512; ++t) {
    float acc = embproj[((size_t)b * 512 + t) * 256 + j];
#pragma unroll
    for (int k = 0; k < 256; ++k) acc += ctx[k] * w[k];
    const float mean = block_sum_256(acc, red) * (1.0f / 256.0f);
    const float d    = acc - mean;
    const float var  = block_sum_256(d * d, red) * (1.0f / 256.0f);
    const float nv   = tanhf(d * rsqrtf(var + LN_EPS) * gj + bj);
    __syncthreads();
    ctx[j] = nv;
    ctxout[((size_t)b * 512 + t) * 256 + j] = (__bf16)nv;
    __syncthreads();
  }
}

// --------------------------- RoPE (in-place on q,k of qkv) ------------------
__global__ void rope_kernel(float* __restrict__ qkv) {
  const int idx = blockIdx.x * blockDim.x + threadIdx.x;  // 4*512*8*8 = 131072
  const int j = idx & 7;
  const int h = (idx >> 3) & 7;
  const int s = (idx >> 6) & 511;
  const int b = idx >> 15;
  if (b >= 4) return;
  const size_t base = ((size_t)b * 512 + s) * 1536 + h * 192;
  const float inv = __powf(10000.0f, -(float)(2 * j) * (1.0f / 16.0f));
  const float ang = (float)s * inv;
  const float c = __cosf(ang), sn = __sinf(ang);
#pragma unroll
  for (int sec = 0; sec <= 64; sec += 64) {  // q then k section
    const float x1 = qkv[base + sec + j];
    const float x2 = qkv[base + sec + j + 8];
    qkv[base + sec + j]     = x1 * c - x2 * sn;
    qkv[base + sec + j + 8] = x2 * c + x1 * sn;
  }
}

// --------------------------- fused causal attention -------------------------
__global__ void __launch_bounds__(32) attn_kernel(const float* __restrict__ qkv,
                                                  __bf16* __restrict__ outb) {
  __shared__ float  sc[16 * 512];   // scores for this q-tile (32 KB)
  __shared__ __bf16 pr[16 * 512];   // un-normalized probs     (16 KB)
  __shared__ float  rinv[16];       // 1/rowsum
  const int qt = blockIdx.x;        // 0..31
  const int h  = blockIdx.y;        // 0..7
  const int b  = blockIdx.z;        // 0..3
  const int lane = threadIdx.x;
  const int n16  = lane & 15;
  const int half = lane >> 4;
  const size_t base_bs = (size_t)b * 512;

  v16bf aq0, aq1;
  {
    const float* qp = qkv + (base_bs + qt * 16 + n16) * 1536 + h * 192;
#pragma unroll
    for (int i = 0; i < 16; ++i) {
      const int kl = (i < 8) ? (half * 8 + i) : (16 + half * 8 + (i - 8));
      aq0[i] = (__bf16)qp[kl];
      aq1[i] = (__bf16)qp[32 + kl];
    }
  }
  for (int idx = lane; idx < 16 * 512; idx += 32) sc[idx] = -1e9f;
  __syncthreads();

  for (int kt = 0; kt <= qt; ++kt) {
    const int key = kt * 16 + n16;
    const float* kp = qkv + (base_bs + key) * 1536 + h * 192 + 64;
    v16bf bk0, bk1;
#pragma unroll
    for (int i = 0; i < 16; ++i) {
      const int kl = (i < 8) ? (half * 8 + i) : (16 + half * 8 + (i - 8));
      bk0[i] = (__bf16)kp[kl];
      bk1[i] = (__bf16)kp[32 + kl];
    }
    v8f s = {0.f, 0.f, 0.f, 0.f, 0.f, 0.f, 0.f, 0.f};
    s = __builtin_amdgcn_wmma_f32_16x16x32_bf16(false, aq0, false, bk0, (short)0, s, false, false);
    s = __builtin_amdgcn_wmma_f32_16x16x32_bf16(false, aq1, false, bk1, (short)0, s, false, false);
#pragma unroll
    for (int r = 0; r < 8; ++r) {
      const int qrow = qt * 16 + half * 8 + r;
      sc[(half * 8 + r) * 512 + key] = (key <= qrow) ? s[r] * 0.125f : -1e9f;
    }
  }
  __syncthreads();

  const int nkeys = qt * 16 + 16;
  if (lane < 16) {
    const int row = lane;
    float mx = -1e30f;
    for (int jj = 0; jj < nkeys; ++jj) mx = fmaxf(mx, sc[row * 512 + jj]);
    float sum = 0.0f;
    for (int jj = 0; jj < 512; ++jj) {
      const float e = (jj < nkeys) ? __expf(sc[row * 512 + jj] - mx) : 0.0f;
      pr[row * 512 + jj] = (__bf16)e;
      sum += e;
    }
    rinv[row] = 1.0f / sum;
  }
  __syncthreads();

  v8f o[4];
  const v8f zero = {0.f, 0.f, 0.f, 0.f, 0.f, 0.f, 0.f, 0.f};
#pragma unroll
  for (int t = 0; t < 4; ++t) o[t] = zero;

  const int kpairs = (qt >> 1) + 1;  // pairs of key tiles (32 keys each)
  for (int kpr = 0; kpr < kpairs; ++kpr) {
    v16bf pa;
#pragma unroll
    for (int i = 0; i < 16; ++i) {
      const int kl = (i < 8) ? (half * 8 + i) : (16 + half * 8 + (i - 8));
      pa[i] = pr[n16 * 512 + kpr * 32 + kl];
    }
#pragma unroll
    for (int t = 0; t < 4; ++t) {
      const int dim = t * 16 + n16;
      v16bf bv;
#pragma unroll
      for (int i = 0; i < 16; ++i) {
        const int kl = (i < 8) ? (half * 8 + i) : (16 + half * 8 + (i - 8));
        const int key = kpr * 32 + kl;
        bv[i] = (__bf16)qkv[(base_bs + key) * 1536 + h * 192 + 128 + dim];
      }
      o[t] = __builtin_amdgcn_wmma_f32_16x16x32_bf16(false, pa, false, bv, (short)0, o[t], false, false);
    }
  }
#pragma unroll
  for (int t = 0; t < 4; ++t) {
#pragma unroll
    for (int r = 0; r < 8; ++r) {
      const int orow = qt * 16 + half * 8 + r;
      outb[(base_bs + orow) * 512 + h * 64 + t * 16 + n16] =
          (__bf16)(o[t][r] * rinv[half * 8 + r]);
    }
  }
}

// --------------------------- residual: x += a + b ---------------------------
__global__ void add3_kernel(float* __restrict__ x, const float* __restrict__ a,
                            const float* __restrict__ bsrc, int n) {
  for (int i = blockIdx.x * blockDim.x + threadIdx.x; i < n;
       i += gridDim.x * blockDim.x)
    x[i] += a[i] + bsrc[i];
}

// ---------------------------------------------------------------------------
extern "C" void kernel_launch(void* const* d_in, const int* in_sizes, int n_in,
                              void* d_out, int out_size, void* d_ws, size_t ws_size,
                              hipStream_t stream) {
  (void)in_sizes; (void)n_in; (void)out_size; (void)ws_size;
  const int*   input_ids    = (const int*)d_in[0];
  const float* prev_context = (const float*)d_in[1];
  const float* embed_W      = (const float*)d_in[2];
  const float* en_g         = (const float*)d_in[3];
  const float* en_b         = (const float*)d_in[4];
  const float* cb_W         = (const float*)d_in[5];
  const float* cb_b         = (const float*)d_in[6];
  const float* cb_g         = (const float*)d_in[7];
  const float* cb_beta      = (const float*)d_in[8];
  const float* cp_W         = (const float*)d_in[9];
  const float* cp_b         = (const float*)d_in[10];
  const float* ln1_g        = (const float*)d_in[11];
  const float* ln1_b        = (const float*)d_in[12];
  const float* ln2_g        = (const float*)d_in[13];
  const float* ln2_b        = (const float*)d_in[14];
  const float* qkv_W        = (const float*)d_in[15];
  const float* qkv_b        = (const float*)d_in[16];
  const float* attn_W       = (const float*)d_in[17];
  const float* attn_b       = (const float*)d_in[18];
  const float* fc_W         = (const float*)d_in[19];
  const float* fc_b         = (const float*)d_in[20];
  const float* fo_W         = (const float*)d_in[21];
  const float* fo_b         = (const float*)d_in[22];
  const float* fln_g        = (const float*)d_in[23];
  const float* fln_b        = (const float*)d_in[24];
  const float* out_W        = (const float*)d_in[25];

  const int M = 2048;  // B*S
  char* ws = (char*)d_ws;
  size_t off = 0;
  auto take = [&](size_t bytes) -> char* {
    char* p = ws + off;
    off = (off + bytes + 255) & ~(size_t)255;
    return p;
  };
  __bf16* tokb     = (__bf16*)take((size_t)M * 512 * 2);
  float*  embproj  = (float*)take((size_t)M * 256 * 4);
  __bf16* ctxb     = (__bf16*)take((size_t)M * 256 * 2);
  float*  xbuf     = (float*)take((size_t)M * 512 * 4);
  __bf16* lnb      = (__bf16*)take((size_t)M * 512 * 2);
  float*  qkv      = (float*)take((size_t)M * 1536 * 4);
  __bf16* attnb    = (__bf16*)take((size_t)M * 512 * 2);
  float*  attnproj = (float*)take((size_t)M * 512 * 4);
  __bf16* mlp1b    = (__bf16*)take((size_t)M * 2048 * 2);
  float*  mlp2     = (float*)take((size_t)M * 512 * 4);
  __bf16* bpack    = (__bf16*)take((size_t)512 * 50304 * 2);

  auto gemm = [&](const __bf16* A, const float* W, const float* bias, void* C,
                  int K, int N, int act, int obf16) {
    pack_b_kernel<<<2048, 256, 0, stream>>>(W, bpack, K, N);
    dim3 grid(N / 64, M / 128);
    if (act == 0 && obf16 == 0)
      gemm_bf16_wmma<0, 0><<<grid, 256, 0, stream>>>(A, bpack, bias, C, M, N, K);
    else
      gemm_bf16_wmma<1, 1><<<grid, 256, 0, stream>>>(A, bpack, bias, C, M, N, K);
  };

  // 1) embedding gather + LN -> bf16 tokens
  embed_ln_kernel<<<M, 256, 0, stream>>>(input_ids, embed_W, en_g, en_b, tokb);
  // 2) hoist emb part of scan: embproj = tok @ cb_W[256:,:] + cb_b
  gemm(tokb, cb_W + 256 * 256, cb_b, embproj, 512, 256, 0, 0);
  // 3) serial context scan (ctx @ W_c in registers) -> bf16 context
  scan_kernel<<<4, 256, 0, stream>>>(cb_W, embproj, prev_context, cb_g, cb_beta, ctxb);
  // 4) x = context @ cp_W + cp_b
  gemm(ctxb, cp_W, cp_b, xbuf, 256, 512, 0, 0);

  for (int l = 0; l < 6; ++l) {
    ln512_kernel<<<M, 256, 0, stream>>>(xbuf, ln1_g + l * 512, ln1_b + l * 512, lnb);
    gemm(lnb, qkv_W + (size_t)l * 512 * 1536, qkv_b + l * 1536, qkv, 512, 1536, 0, 0);
    rope_kernel<<<512, 256, 0, stream>>>(qkv);
    attn_kernel<<<dim3(32, 8, 4), 32, 0, stream>>>(qkv, attnb);
    gemm(attnb, attn_W + (size_t)l * 512 * 512, attn_b + l * 512, attnproj, 512, 512, 0, 0);
    ln512_kernel<<<M, 256, 0, stream>>>(xbuf, ln2_g + l * 512, ln2_b + l * 512, lnb);
    gemm(lnb, fc_W + (size_t)l * 512 * 2048, fc_b + l * 2048, mlp1b, 512, 2048, 1, 1);
    gemm(mlp1b, fo_W + (size_t)l * 2048 * 512, fo_b + l * 512, mlp2, 2048, 512, 0, 0);
    add3_kernel<<<1024, 256, 0, stream>>>(xbuf, attnproj, mlp2, M * 512);
  }

  // final LN + vocab projection (no bias)
  ln512_kernel<<<M, 256, 0, stream>>>(xbuf, fln_g, fln_b, lnb);
  gemm(lnb, out_W, nullptr, d_out, 512, 50304, 0, 0);
}
